// SpGAT_2_73065983639801
// MI455X (gfx1250) — matrix operand order repaired and checked by reading, not compiled, not run on hardware
//
#include <hip/hip_runtime.h>
#include <hip/hip_bf16.h>

// ---------------------------------------------------------------------------
// SpGAT_2 for MI455X (gfx1250): bf16 WMMA GEMMs (register-blocked 16x64 per
// wave) + wave-per-edge f32 atomic scatter for the segment sums. Edge scatter
// is the roofline limiter (~1.6GB random gather+atomic traffic); the GEMMs
// (26 GFLOP bf16) ride v_wmma_f32_16x16x32_bf16.
// ---------------------------------------------------------------------------

typedef __bf16 bf16;
typedef bf16  v16bf __attribute__((ext_vector_type(16)));
typedef bf16  v8bf  __attribute__((ext_vector_type(8)));
typedef float v8f   __attribute__((ext_vector_type(8)));

#define NN     50000
#define EE     800000
#define NFEAT  512
#define NHID   128
#define KFACT  4
#define NCLASS 64
#define ALPHA  0.2f
#define EPSV   1e-16f

__device__ __forceinline__ bf16 f2bf(float f) {
  union { float f; unsigned u; } v; v.f = f;
  unsigned r = v.u + 0x7FFFu + ((v.u >> 16) & 1u);      // round-to-nearest-even
  unsigned short h = (unsigned short)(r >> 16);
  return __builtin_bit_cast(bf16, h);
}

__device__ __forceinline__ float wred_sum(float v) {
  #pragma unroll
  for (int m = 16; m > 0; m >>= 1) v += __shfl_xor(v, m, 32);
  return v;
}
__device__ __forceinline__ float wred_max(float v) {
  #pragma unroll
  for (int m = 16; m > 0; m >>= 1) v = fmaxf(v, __shfl_xor(v, m, 32));
  return v;
}
__device__ __forceinline__ float eluf(float x) {
  return x > 0.f ? x : (__expf(x) - 1.f);
}

// ---------------------------- init / utility -------------------------------

__global__ void zero_f32(float* __restrict__ p, size_t n) {
  size_t i = (size_t)blockIdx.x * blockDim.x + threadIdx.x;
  size_t stride = (size_t)gridDim.x * blockDim.x;
  for (; i < n; i += stride) p[i] = 0.f;
}

__global__ void conv_x_bf16(const float* __restrict__ x, bf16* __restrict__ xb, size_t n) {
  size_t i = (size_t)blockIdx.x * blockDim.x + threadIdx.x;
  if (i < n) xb[i] = f2bf(x[i]);
}

// Wt[c][f] = W_ks[k][f][j], c = k*128+j  -> (512 x 512), row-major over f
__global__ void conv_wks(const float* __restrict__ Wks, bf16* __restrict__ Wt) {
  int i = blockIdx.x * blockDim.x + threadIdx.x;     // 512*512
  if (i >= NFEAT * NFEAT) return;
  int c = i >> 9, f = i & 511;
  int k = c >> 7, j = c & 127;
  Wt[(size_t)c * NFEAT + f] = f2bf(Wks[((size_t)k * NFEAT + f) * NHID + j]);
}

// Wot[c][f] = W_o[f][c]  -> (64 x 512)
__global__ void conv_wo(const float* __restrict__ Wo, bf16* __restrict__ Wot) {
  int i = blockIdx.x * blockDim.x + threadIdx.x;     // 64*512
  if (i >= NCLASS * NFEAT) return;
  int c = i >> 9, f = i & 511;
  Wot[(size_t)c * NFEAT + f] = f2bf(Wo[(size_t)f * NCLASS + c]);
}

// ------------------------------ WMMA GEMM ----------------------------------
// C(M x Nc) = A(M x Kd, bf16 row-major) @ Bt(Nc x Kd, bf16 row-major)^T
// Register-blocked: each wave computes a 16x64 strip (4 accumulators), so one
// A fragment feeds 4 wmma per k-step. 4 waves / block cover 64 rows.
// Requires: Kd % 32 == 0, M % 16 == 0, Nc % 64 == 0.
__global__ __launch_bounds__(128) void wmma_gemm_bf16(
    const bf16* __restrict__ A, const bf16* __restrict__ Bt,
    float* __restrict__ C, int M, int Nc, int Kd) {
  const int lane  = threadIdx.x & 31;
  const int wave  = threadIdx.x >> 5;
  const int tileM = blockIdx.x * 4 + wave;
  if (tileM * 16 >= M) return;                        // wave-uniform guard
  const int colBase = blockIdx.y * 64;                // 4 col tiles per wave

  const int rc    = lane & 15;                        // row/col within tile
  const int half  = lane >> 4;                        // 0 or 1
  const size_t arow = (size_t)(tileM * 16 + rc) * Kd;
  size_t brow[4];
  #pragma unroll
  for (int nt = 0; nt < 4; ++nt)
    brow[nt] = (size_t)(colBase + nt * 16 + rc) * Kd;

  v8f acc[4] = {};
  for (int kk = 0; kk < Kd; kk += 32) {
    // A 16x32 fragment: lanes 0-15 K {0..7,16..23}, lanes 16-31 {8..15,24..31}
    v8bf alo = *(const v8bf*)(A + arow + kk + half * 8);
    v8bf ahi = *(const v8bf*)(A + arow + kk + 16 + half * 8);
    v16bf a;
    #pragma unroll
    for (int i = 0; i < 8; ++i) { a[i] = alo[i]; a[8 + i] = ahi[i]; }
    #pragma unroll
    for (int nt = 0; nt < 4; ++nt) {
      // B 32x16 fragment from transposed Bt: lanes 0-15 K {0..7,16..23}, lanes 16-31 {8..15,24..31}
      v8bf blo = *(const v8bf*)(Bt + brow[nt] + kk + half * 8);
      v8bf bhi = *(const v8bf*)(Bt + brow[nt] + kk + 16 + half * 8);
      v16bf b;
      #pragma unroll
      for (int i = 0; i < 8; ++i) { b[i] = blo[i]; b[8 + i] = bhi[i]; }
      acc[nt] = __builtin_amdgcn_wmma_f32_16x16x32_bf16(
          false, a, false, b, (short)0, acc[nt], false, false);
    }
  }
  const int orow = tileM * 16 + half * 8;
  #pragma unroll
  for (int nt = 0; nt < 4; ++nt) {
    const int ocol = colBase + nt * 16 + rc;
    #pragma unroll
    for (int r = 0; r < 8; ++r)
      C[(size_t)(orow + r) * Nc + ocol] = acc[nt][r];
  }
}

// -------------------------- per-node score dots ----------------------------
// s_src[n][k] = h[n]·a_k[k][0:512]; s_dst[n][k] = h[n]·a_k[k][512:1024]
// s1[n][k] = h_k[n]·a_att[k][0:128]; s2[n][k] = h_k[n]·a_att[k][128:256]
__global__ __launch_bounds__(256) void node_scores(
    const float* __restrict__ h, const float* __restrict__ a_k,
    const float* __restrict__ a_att,
    float4* __restrict__ s_src, float4* __restrict__ s_dst,
    float4* __restrict__ s1v, float4* __restrict__ s2v) {
  const int lane = threadIdx.x & 31;
  const int n = blockIdx.x * 8 + (threadIdx.x >> 5);
  if (n >= NN) return;
  float ss[4] = {0,0,0,0}, sd[4] = {0,0,0,0}, s1a[4] = {0,0,0,0}, s2a[4] = {0,0,0,0};
  #pragma unroll 4
  for (int t = 0; t < 16; ++t) {
    int f = lane + t * 32;
    float hv = h[(size_t)n * NFEAT + f];
    #pragma unroll
    for (int k = 0; k < 4; ++k) {
      ss[k] += hv * a_k[k * 1024 + f];
      sd[k] += hv * a_k[k * 1024 + 512 + f];
    }
    int k = f >> 7, j = f & 127;
    s1a[k] += hv * a_att[k * 256 + j];
    s2a[k] += hv * a_att[k * 256 + 128 + j];
  }
  #pragma unroll
  for (int k = 0; k < 4; ++k) {
    ss[k] = wred_sum(ss[k]);  sd[k] = wred_sum(sd[k]);
    s1a[k] = wred_sum(s1a[k]); s2a[k] = wred_sum(s2a[k]);
  }
  if (lane == 0) {
    s_src[n] = make_float4(ss[0], ss[1], ss[2], ss[3]);
    s_dst[n] = make_float4(sd[0], sd[1], sd[2], sd[3]);
    s1v[n]   = make_float4(s1a[0], s1a[1], s1a[2], s1a[3]);
    s2v[n]   = make_float4(s2a[0], s2a[1], s2a[2], s2a[3]);
  }
}

// ------------------- edge kernel: routing + factor scatter -----------------
// One wave per edge; top-2 of softmax over K=4 picks active factors; gathers
// h[dst] rows and atomic-adds into hp[src] (coalesced: lane f = lane+32q).
__global__ __launch_bounds__(256) void edge_factor(
    const int* __restrict__ src, const int* __restrict__ dst,
    const float4* __restrict__ s_src, const float4* __restrict__ s_dst,
    const float4* __restrict__ s1v, const float4* __restrict__ s2v,
    const float* __restrict__ h, float* __restrict__ hp,
    float* __restrict__ rowsum, float* __restrict__ loss_acc) {
  __shared__ float lsum[8];
  const int lane = threadIdx.x & 31;
  const int wid  = threadIdx.x >> 5;
  if (threadIdx.x < 8) lsum[threadIdx.x] = 0.f;
  __syncthreads();
  const int e = blockIdx.x * 8 + wid;
  if (e < EE) {
    const int s = src[e], d = dst[e];
    float4 as = s_src[s], ad = s_dst[d];
    float l[4] = {as.x + ad.x, as.y + ad.y, as.z + ad.z, as.w + ad.w};
    float m = fmaxf(fmaxf(l[0], l[1]), fmaxf(l[2], l[3]));
    float p[4], psum = 0.f;
    #pragma unroll
    for (int k = 0; k < 4; ++k) { p[k] = __expf(l[k] - m); psum += p[k]; }
    float inv = 1.f / psum;
    int i1 = 0;
    #pragma unroll
    for (int k = 1; k < 4; ++k) if (p[k] > p[i1]) i1 = k;
    int i2 = (i1 == 0) ? 1 : 0;
    #pragma unroll
    for (int k = 0; k < 4; ++k) if (k != i1 && p[k] > p[i2]) i2 = k;
    if (lane == 0) lsum[wid] = 1.f - (p[i1] + p[i2]) * inv;

    float4 s1s = s1v[s], s2d = s2v[d];
    float s1arr[4] = {s1s.x, s1s.y, s1s.z, s1s.w};
    float s2arr[4] = {s2d.x, s2d.y, s2d.z, s2d.w};
    #pragma unroll
    for (int t = 0; t < 2; ++t) {
      const int k = t ? i2 : i1;
      float sc = s1arr[k] + s2arr[k];
      float lr = sc > 0.f ? sc : ALPHA * sc;
      float ev = __expf(-lr);
      if (lane == 0) atomicAdd(&rowsum[(size_t)s * 4 + k], ev);
      const float* hrow  = h  + (size_t)d * NFEAT + k * NHID;
      float*       hprow = hp + (size_t)s * NFEAT + k * NHID;
      #pragma unroll
      for (int q = 0; q < 4; ++q) {
        int f = lane + q * 32;
        atomicAdd(&hprow[f], ev * hrow[f]);
      }
    }
  }
  __syncthreads();
  if (threadIdx.x == 0) {
    float t = 0.f;
    #pragma unroll
    for (int i = 0; i < 8; ++i) t += lsum[i];
    atomicAdd(loss_acc, t);
  }
}

// ----------------- normalize + elu + concat -> bf16 for GEMM2 --------------
__global__ void normalize_cat(const float* __restrict__ hp,
                              const float* __restrict__ rowsum,
                              bf16* __restrict__ outs) {
  size_t i = (size_t)blockIdx.x * blockDim.x + threadIdx.x;
  if (i >= (size_t)NN * NFEAT) return;
  int n = (int)(i >> 9), f = (int)(i & 511), k = f >> 7;
  float v = hp[i] / (rowsum[(size_t)n * 4 + k] + EPSV);
  outs[i] = f2bf(eluf(v));
}

// ----------------------- final layer node scores ---------------------------
__global__ __launch_bounds__(256) void node_scores_out(
    const float* __restrict__ xo, const float* __restrict__ a_out,
    float* __restrict__ s1o, float* __restrict__ s2o) {
  const int lane = threadIdx.x & 31;
  const int n = blockIdx.x * 8 + (threadIdx.x >> 5);
  if (n >= NN) return;
  float a1 = 0.f, a2 = 0.f;
  #pragma unroll
  for (int t = 0; t < 2; ++t) {
    int c = lane + t * 32;
    float v = xo[(size_t)n * NCLASS + c];
    a1 += v * a_out[c];
    a2 += v * a_out[NCLASS + c];
  }
  a1 = wred_sum(a1); a2 = wred_sum(a2);
  if (lane == 0) { s1o[n] = a1; s2o[n] = a2; }
}

// ------------------------- final edge scatter ------------------------------
__global__ __launch_bounds__(256) void edge_final(
    const int* __restrict__ src, const int* __restrict__ dst,
    const float* __restrict__ s1o, const float* __restrict__ s2o,
    const float* __restrict__ xo, float* __restrict__ hp2,
    float* __restrict__ rowsum2) {
  const int lane = threadIdx.x & 31;
  const int e = blockIdx.x * 8 + (threadIdx.x >> 5);
  if (e >= EE) return;
  const int s = src[e], d = dst[e];
  float sc = s1o[s] + s2o[d];
  float lr = sc > 0.f ? sc : ALPHA * sc;
  float ev = __expf(-lr);
  if (lane == 0) atomicAdd(&rowsum2[s], ev);
  #pragma unroll
  for (int t = 0; t < 2; ++t) {
    int f = lane + t * 32;
    atomicAdd(&hp2[(size_t)s * NCLASS + f], ev * xo[(size_t)d * NCLASS + f]);
  }
}

// --------------------- normalize + elu + log_softmax -----------------------
__global__ __launch_bounds__(256) void finalize(
    const float* __restrict__ hp2, const float* __restrict__ rowsum2,
    float* __restrict__ out) {
  const int lane = threadIdx.x & 31;
  const int n = blockIdx.x * 8 + (threadIdx.x >> 5);
  if (n >= NN) return;
  float inv = 1.f / (rowsum2[n] + EPSV);
  float v0 = eluf(hp2[(size_t)n * NCLASS + lane] * inv);
  float v1 = eluf(hp2[(size_t)n * NCLASS + lane + 32] * inv);
  float m = wred_max(fmaxf(v0, v1));
  float se = wred_sum(__expf(v0 - m) + __expf(v1 - m));
  float lse = m + __logf(se);
  out[(size_t)n * NCLASS + lane]      = v0 - lse;
  out[(size_t)n * NCLASS + lane + 32] = v1 - lse;
}

__global__ void write_loss(const float* __restrict__ loss_acc, float* __restrict__ out) {
  if (threadIdx.x == 0 && blockIdx.x == 0)
    out[(size_t)NN * NCLASS] = loss_acc[0] / (float)EE;
}

// ---------------------------------------------------------------------------

extern "C" void kernel_launch(void* const* d_in, const int* in_sizes, int n_in,
                              void* d_out, int out_size, void* d_ws, size_t ws_size,
                              hipStream_t stream) {
  (void)in_sizes; (void)n_in; (void)out_size; (void)ws_size;
  const float* x     = (const float*)d_in[0];
  const int*   edge  = (const int*)d_in[1];
  const float* Wks   = (const float*)d_in[2];
  const float* a_k   = (const float*)d_in[3];
  const float* Wo    = (const float*)d_in[4];
  const float* a_att = (const float*)d_in[5];
  const float* a_out = (const float*)d_in[6];
  const int* src = edge;
  const int* dst = edge + EE;
  float* out = (float*)d_out;

  // -------- workspace layout (256B aligned slices) --------
  char* ws = (char*)d_ws;
  size_t off = 0;
  auto alloc = [&](size_t bytes) -> void* {
    void* p = ws + off;
    off = (off + bytes + 255) & ~(size_t)255;
    return p;
  };
  float* h       = (float*)alloc((size_t)NN * NFEAT * 4);
  // accumulators: contiguous so one zero pass covers everything
  float* hp      = (float*)alloc((size_t)NN * NFEAT * 4);
  float* rowsum  = (float*)alloc((size_t)NN * 4 * 4);
  float* hp2     = (float*)alloc((size_t)NN * NCLASS * 4);
  float* rowsum2 = (float*)alloc((size_t)NN * 4);
  float* lossacc = (float*)alloc(256);
  size_t zero_count = ((size_t)((char*)lossacc - (char*)hp)) / 4 + 64;
  float* s_src   = (float*)alloc((size_t)NN * 4 * 4);
  float* s_dst   = (float*)alloc((size_t)NN * 4 * 4);
  float* s1v     = (float*)alloc((size_t)NN * 4 * 4);
  float* s2v     = (float*)alloc((size_t)NN * 4 * 4);
  float* xo      = (float*)alloc((size_t)NN * NCLASS * 4);
  float* s1o     = (float*)alloc((size_t)NN * 4);
  float* s2o     = (float*)alloc((size_t)NN * 4);
  bf16*  xbf     = (bf16*)alloc((size_t)NN * NFEAT * 2);   // later reused as outs_bf16
  bf16*  Wt      = (bf16*)alloc((size_t)NFEAT * NFEAT * 2);
  bf16*  Wot     = (bf16*)alloc((size_t)NCLASS * NFEAT * 2);

  // 1. zero accumulators (hp, rowsum, hp2, rowsum2, loss)
  zero_f32<<<2048, 256, 0, stream>>>(hp, zero_count);

  // 2. dtype conversions / weight transposes
  conv_x_bf16<<<(int)(((size_t)NN * NFEAT + 255) / 256), 256, 0, stream>>>(x, xbf, (size_t)NN * NFEAT);
  conv_wks<<<(NFEAT * NFEAT + 255) / 256, 256, 0, stream>>>(Wks, Wt);
  conv_wo<<<(NCLASS * NFEAT + 255) / 256, 256, 0, stream>>>(Wo, Wot);

  // 3. h = x @ [W_ks[0]|...|W_ks[3]]  (50000x512x512, WMMA bf16)
  {
    dim3 grid((NN / 16 + 3) / 4, NFEAT / 64);
    wmma_gemm_bf16<<<grid, 128, 0, stream>>>(xbf, Wt, h, NN, NFEAT, NFEAT);
  }

  // 4. per-node routing & attention scores
  node_scores<<<(NN + 7) / 8, 256, 0, stream>>>(h, a_k, a_att,
      (float4*)s_src, (float4*)s_dst, (float4*)s1v, (float4*)s2v);

  // 5. edge routing (softmax-K, top-2) + masked attention scatter
  edge_factor<<<(EE + 7) / 8, 256, 0, stream>>>(src, dst,
      (const float4*)s_src, (const float4*)s_dst,
      (const float4*)s1v, (const float4*)s2v, h, hp, rowsum, lossacc);

  // 6. normalize, elu, concat -> bf16 (reuse xbf buffer as outs)
  normalize_cat<<<(int)(((size_t)NN * NFEAT + 255) / 256), 256, 0, stream>>>(hp, rowsum, xbf);

  // 7. xo = outs @ W_o  (50000x512x64, WMMA bf16; one 16x64 strip per wave)
  {
    dim3 grid((NN / 16 + 3) / 4, NCLASS / 64);
    wmma_gemm_bf16<<<grid, 128, 0, stream>>>(xbf, Wot, xo, NN, NCLASS, NFEAT);
  }

  // 8. final-layer node scores
  node_scores_out<<<(NN + 7) / 8, 256, 0, stream>>>(xo, a_out, s1o, s2o);

  // 9. final unmasked attention scatter
  edge_final<<<(EE + 7) / 8, 256, 0, stream>>>(src, dst, s1o, s2o, xo, hp2, rowsum2);

  // 10. normalize, elu, log_softmax -> d_out[0 : N*64]
  finalize<<<(NN + 7) / 8, 256, 0, stream>>>(hp2, rowsum2, out);

  // 11. att_loss -> d_out[N*64]
  write_loss<<<1, 32, 0, stream>>>(lossacc, out);
}
